// QGCL_14516989461112
// MI455X (gfx1250) — matrix operand order repaired and checked by reading, not compiled, not run on hardware
//
#include <hip/hip_runtime.h>
#include <hip/hip_bf16.h>

typedef __attribute__((ext_vector_type(16))) _Float16 v16h;
typedef __attribute__((ext_vector_type(8)))  _Float16 v8h;
typedef __attribute__((ext_vector_type(8)))  float    v8f;

// ---------------------------------------------------------------------------
// WMMA helpers (CDNA5 wave32, V_WMMA_F32_16X16X32_F16)
// ---------------------------------------------------------------------------
__device__ __forceinline__ v8f wmma_f16(v16h a, v16h b, v8f c) {
  // 8 args: (neg_a, A, neg_b, B, c_mod, C, reuse_a, reuse_b)
  return __builtin_amdgcn_wmma_f32_16x16x32_f16(false, a, false, b, (short)0, c,
                                                false, false);
}

// A fragment (16x32 f16, ISA 7.12.2 layout) from a row-major f16 LDS tile.
// lanes 0-15: rows 0-15, halves 0-7 = K[kbase..kbase+7], halves 8-15 = K[kbase+16..+23]
// lanes 16-31: same rows, K offsets +8.
__device__ __forceinline__ v16h load_a_frag(const _Float16* tile, int strideHalves,
                                            int lane, int kbase) {
  int row = lane & 15, hi = lane >> 4;
  const _Float16* p0 = tile + row * strideHalves + kbase + 8 * hi;
  v8h lo = *(const v8h*)p0;
  v8h hq = *(const v8h*)(p0 + 16);
  v16h r;
#pragma unroll
  for (int i = 0; i < 8; ++i) { r[i] = lo[i]; r[8 + i] = hq[i]; }
  return r;
}

// fast silu: x * rcp(1 + exp(-x))  (v_exp_f32 + v_rcp_f32, no IEEE div sequence)
__device__ __forceinline__ float silu_f(float x) {
  return x * __builtin_amdgcn_rcpf(1.0f + __expf(-x));
}

// ---------------------------------------------------------------------------
// Pack a [K][128] f32 weight into f16 B-fragments.
// B fragment layout (32x16, per ISA dense-B / SWMMAC-B pattern): lane holds
// column N = ntile*16 + (lane&15); half v holds K = kstep*32 + 8*(lane>>4) + (v<8 ? v : v+8).
// Stored lane-major: dst[(ntile*KSTEPS+kstep)*512 + lane*16 + v]
// ---------------------------------------------------------------------------
__global__ void pack_b_frags(const float* __restrict__ src, _Float16* __restrict__ dst,
                             int Ktot) {
  int idx = blockIdx.x * blockDim.x + threadIdx.x;
  int total = Ktot * 128;
  if (idx >= total) return;
  int v = idx & 15;
  int lane = (idx >> 4) & 31;
  int fi = idx >> 9;
  int ksteps = Ktot >> 5;
  int ntile = fi / ksteps;
  int kstep = fi - ntile * ksteps;
  int K = kstep * 32 + 8 * (lane >> 4) + (v < 8 ? v : v + 8);
  int N = ntile * 16 + (lane & 15);
  dst[idx] = (_Float16)src[(size_t)K * 128 + N];
}

__global__ void zero_f32(float* __restrict__ p, long n) {
  long i = (long)blockIdx.x * blockDim.x + threadIdx.x;
  if (i < n) p[i] = 0.0f;
}

// ---------------------------------------------------------------------------
// Cayley: per layer j (blockIdx.x), build M = A + A^H + iI, Gauss-Jordan invert
// (no pivoting; M ~ iI + small hermitian, well conditioned), then
// Qt[k][d] = ((A_h - iI) @ inv(M))[d][k], split into real/imag f32 planes.
// Single workgroup per layer; matrix lives in global workspace.
// ---------------------------------------------------------------------------
__global__ __launch_bounds__(256) void cayley_kernel(
    const float* __restrict__ A_real, const float* __restrict__ A_imag,
    float* __restrict__ augAll, float* __restrict__ Qt) {
  const int j = blockIdx.x;
  const int tid = threadIdx.x;
  const float* Ar = A_real + (size_t)j * 128 * 128;
  const float* Ai = A_imag + (size_t)j * 128 * 128;
  float* M = augAll + (size_t)j * 128 * 256 * 2;  // interleaved complex [128][256]

  __shared__ float pivr, pivi;
  __shared__ float facr[128], faci[128];
  __shared__ float prow[256 * 2];

  // init augmented [M | I]
  for (int i = tid; i < 128 * 256; i += 256) {
    int r = i >> 8, c = i & 255;
    float re, im;
    if (c < 128) {
      re = Ar[r * 128 + c] + Ar[c * 128 + r];
      im = Ai[r * 128 + c] - Ai[c * 128 + r];
      if (c == r) im += 1.0f;
    } else {
      re = ((c - 128) == r) ? 1.0f : 0.0f;
      im = 0.0f;
    }
    M[i * 2] = re;
    M[i * 2 + 1] = im;
  }
  __syncthreads();

  for (int p = 0; p < 128; ++p) {
    if (tid == 0) {
      float zr = M[(p * 256 + p) * 2], zi = M[(p * 256 + p) * 2 + 1];
      float d = zr * zr + zi * zi;
      pivr = zr / d;
      pivi = -zi / d;
    }
    __syncthreads();
    {  // scale pivot row; cache it in LDS
      int c = tid;
      float xr = M[(p * 256 + c) * 2], xi = M[(p * 256 + c) * 2 + 1];
      float nr = xr * pivr - xi * pivi;
      float ni = xr * pivi + xi * pivr;
      M[(p * 256 + c) * 2] = nr;
      M[(p * 256 + c) * 2 + 1] = ni;
      prow[c * 2] = nr;
      prow[c * 2 + 1] = ni;
    }
    __syncthreads();
    if (tid < 128) {
      facr[tid] = M[(tid * 256 + p) * 2];
      faci[tid] = M[(tid * 256 + p) * 2 + 1];
    }
    __syncthreads();
    for (int i = tid; i < 128 * 256; i += 256) {
      int r = i >> 8, c = i & 255;
      if (r == p) continue;
      float fr = facr[r], fi = faci[r];
      float pr = prow[c * 2], pi = prow[c * 2 + 1];
      M[i * 2] -= fr * pr - fi * pi;
      M[i * 2 + 1] -= fr * pi + fi * pr;
    }
    __syncthreads();
  }

  // Q = (A_h - iI) @ invM ; store transposed as [K][N]: Qt[k*128+d] = Q[d][k]
  float* Qtr = Qt + (size_t)(j * 2 + 0) * 128 * 128;
  float* Qti = Qt + (size_t)(j * 2 + 1) * 128 * 128;
  for (int i = tid; i < 128 * 128; i += 256) {
    int d = i >> 7, k = i & 127;
    float sr = 0.0f, si = 0.0f;
    for (int s = 0; s < 128; ++s) {
      float br = Ar[d * 128 + s] + Ar[s * 128 + d];
      float bi = Ai[d * 128 + s] - Ai[s * 128 + d];
      if (s == d) bi -= 1.0f;
      float mr = M[(s * 256 + 128 + k) * 2], mi = M[(s * 256 + 128 + k) * 2 + 1];
      sr += br * mr - bi * mi;
      si += br * mi + bi * mr;
    }
    Qtr[k * 128 + d] = sr;
    Qti[k * 128 + d] = si;
  }
}

// ---------------------------------------------------------------------------
// Edge MLP: per 16-edge tile, e=[h[row]||h[col]] -> silu(e@W_e1+b1) ->
// silu(.@W_e2+b2) -> edge_feat + atomic scatter into agg.
// 8 waves/block; wave w owns output columns [16w,16w+16) with its W fragments
// register-resident across the whole grid-stride loop. A fragments are
// double-buffered so ds_loads overlap the WMMA pipe.
// ---------------------------------------------------------------------------
__global__ __launch_bounds__(256) void edge_kernel(
    const float* __restrict__ h, const int* __restrict__ ei,
    const _Float16* __restrict__ We1f, const _Float16* __restrict__ We2f,
    const float* __restrict__ b_e1, const float* __restrict__ b_e2,
    float* __restrict__ agg, float* __restrict__ edge_feat, int N, int E) {
  __shared__ _Float16 A1[16][264];  // 16 edges x 256 (padded)
  __shared__ _Float16 A2[16][136];  // 16 edges x 128 (padded)

  const int tid = threadIdx.x;
  const int lane = tid & 31;
  const int wave = tid >> 5;
  const int hi = lane >> 4;
  const int nn = wave * 16 + (lane & 15);

  v16h L1B[8], L2B[4];
#pragma unroll
  for (int ks = 0; ks < 8; ++ks)
    L1B[ks] = *(const v16h*)(We1f + ((size_t)(wave * 8 + ks) * 512 + lane * 16));
#pragma unroll
  for (int ks = 0; ks < 4; ++ks)
    L2B[ks] = *(const v16h*)(We2f + ((size_t)(wave * 4 + ks) * 512 + lane * 16));

  const float b1 = b_e1[nn];
  const float b2 = b_e2[nn];
  const int nTiles = (E + 15) >> 4;

  for (int t = blockIdx.x; t < nTiles; t += gridDim.x) {
    int tn = t + gridDim.x;
    if (tn < nTiles && tid == 0) {  // global_prefetch_b8 next tile's indices
      __builtin_prefetch(ei + tn * 16, 0, 1);
      __builtin_prefetch(ei + E + tn * 16, 0, 1);
    }
    // stage A1 = f16([h[row] || h[col]]): 4 b128 loads in flight, then convert
    {
      float4 tmp[4];
#pragma unroll
      for (int u = 0; u < 4; ++u) {
        int i = tid + u * 256;
        int el = i >> 6, c4 = i & 63;
        int e = t * 16 + el;
        tmp[u] = make_float4(0.f, 0.f, 0.f, 0.f);
        if (e < E) {
          int r = (c4 < 32) ? ei[e] : ei[E + e];
          int cc = (c4 < 32) ? c4 : (c4 - 32);
          tmp[u] = ((const float4*)(h + (size_t)r * 128))[cc];
        }
      }
#pragma unroll
      for (int u = 0; u < 4; ++u) {
        int i = tid + u * 256;
        int el = i >> 6, c4 = i & 63;
        _Float16* dst = &A1[el][c4 * 4];
        dst[0] = (_Float16)tmp[u].x; dst[1] = (_Float16)tmp[u].y;
        dst[2] = (_Float16)tmp[u].z; dst[3] = (_Float16)tmp[u].w;
      }
    }
    __syncthreads();

    // layer 1: K=256 (8 wmma), double-buffered A fragments
    v8f acc;
#pragma unroll
    for (int i = 0; i < 8; ++i) acc[i] = b1;
    {
      v16h a = load_a_frag(&A1[0][0], 264, lane, 0);
#pragma unroll
      for (int ks = 0; ks < 8; ++ks) {
        v16h cur = a;
        if (ks < 7) a = load_a_frag(&A1[0][0], 264, lane, (ks + 1) * 32);
        acc = wmma_f16(cur, L1B[ks], acc);
      }
    }
#pragma unroll
    for (int r = 0; r < 8; ++r)
      A2[r + 8 * hi][nn] = (_Float16)silu_f(acc[r]);
    __syncthreads();

    // layer 2: K=128 (4 wmma)
    v8f acc2;
#pragma unroll
    for (int i = 0; i < 8; ++i) acc2[i] = b2;
    {
      v16h a = load_a_frag(&A2[0][0], 136, lane, 0);
#pragma unroll
      for (int ks = 0; ks < 4; ++ks) {
        v16h cur = a;
        if (ks < 3) a = load_a_frag(&A2[0][0], 136, lane, (ks + 1) * 32);
        acc2 = wmma_f16(cur, L2B[ks], acc2);
      }
    }
#pragma unroll
    for (int r = 0; r < 8; ++r) {
      float vv = silu_f(acc2[r]);
      int m = r + 8 * hi;
      int e = t * 16 + m;
      if (e < E) {
        edge_feat[(size_t)e * 128 + nn] = vv;
        atomicAdd(&agg[(size_t)ei[e] * 128 + nn], vv);
      }
    }
    __syncthreads();
  }
}

// ---------------------------------------------------------------------------
// Node pipeline: encoder GEMM -> normalize -> 2x (complex GEMM + 1q gates +
// CNOT permute) -> decoder GEMM + residual. 16 nodes per block iteration.
// ---------------------------------------------------------------------------
__global__ __launch_bounds__(256) void node_kernel(
    const float* __restrict__ h, const float* __restrict__ agg,
    const _Float16* __restrict__ Wencf, const _Float16* __restrict__ Wdecf,
    const _Float16* __restrict__ Qf, const float* __restrict__ b_enc,
    const float* __restrict__ b_dec, const float* __restrict__ coeffs,
    float* __restrict__ out, int N) {
  __shared__ _Float16 Ain[16][264];
  __shared__ float Qm[16][128];
  __shared__ float PR[16][128], PI[16][128];
  __shared__ float QR[16][128], QI[16][128];
  __shared__ _Float16 FR[16][136], FI[16][136];
  __shared__ float part[16][16];
  __shared__ float norms[16];

  const int tid = threadIdx.x;
  const int lane = tid & 31;
  const int wave = tid >> 5;
  const int hi = lane >> 4;
  const int nn = wave * 16 + (lane & 15);

  const float benc = b_enc[nn];
  const float bdec = b_dec[nn];
  const int nTiles = (N + 15) >> 4;

  for (int t = blockIdx.x; t < nTiles; t += gridDim.x) {
    // stage [h || agg] as f16: 4 b128 loads in flight, then convert
    {
      float4 tmp[4];
#pragma unroll
      for (int u = 0; u < 4; ++u) {
        int i = tid + u * 256;
        int nl = i >> 6, c4 = i & 63;
        int node = t * 16 + nl;
        tmp[u] = make_float4(0.f, 0.f, 0.f, 0.f);
        if (node < N) {
          const float* src = (c4 < 32) ? (h + (size_t)node * 128) : (agg + (size_t)node * 128);
          int cc = (c4 < 32) ? c4 : (c4 - 32);
          tmp[u] = ((const float4*)src)[cc];
        }
      }
#pragma unroll
      for (int u = 0; u < 4; ++u) {
        int i = tid + u * 256;
        int nl = i >> 6, c4 = i & 63;
        _Float16* dst = &Ain[nl][c4 * 4];
        dst[0] = (_Float16)tmp[u].x; dst[1] = (_Float16)tmp[u].y;
        dst[2] = (_Float16)tmp[u].z; dst[3] = (_Float16)tmp[u].w;
      }
    }
    __syncthreads();

    // encoder: q_input = [h||agg] @ W_enc + b_enc
    v8f acc;
#pragma unroll
    for (int i = 0; i < 8; ++i) acc[i] = benc;
    {
      v16h a = load_a_frag(&Ain[0][0], 264, lane, 0);
#pragma unroll
      for (int ks = 0; ks < 8; ++ks) {
        v16h cur = a;
        if (ks < 7) a = load_a_frag(&Ain[0][0], 264, lane, (ks + 1) * 32);
        v16h b = *(const v16h*)(Wencf + ((size_t)(wave * 8 + ks) * 512 + lane * 16));
        acc = wmma_f16(cur, b, acc);
      }
    }
#pragma unroll
    for (int r = 0; r < 8; ++r) Qm[r + 8 * hi][nn] = acc[r];
    __syncthreads();

    // row norms (two-stage parallel reduction), psi init
    {
      int rm = tid >> 4, seg = tid & 15;
      float s = 0.f;
#pragma unroll
      for (int k = 0; k < 8; ++k) { float x = Qm[rm][seg * 8 + k]; s += x * x; }
      part[rm][seg] = s;
    }
    __syncthreads();
    if (tid < 16) {
      float s = 1e-12f;
#pragma unroll
      for (int k = 0; k < 16; ++k) s += part[tid][k];
      norms[tid] = 1.0f / sqrtf(s);
    }
    __syncthreads();
    for (int i = tid; i < 2048; i += 256) {
      int m = i >> 7, k = i & 127;
      PR[m][k] = Qm[m][k] * norms[m];
      PI[m][k] = 0.0f;
    }
    __syncthreads();

    for (int j = 0; j < 2; ++j) {
      // f16 staging of psi
      for (int i = tid; i < 2048; i += 256) {
        int m = i >> 7, k = i & 127;
        FR[m][k] = (_Float16)PR[m][k];
        FI[m][k] = (_Float16)PI[m][k];
      }
      __syncthreads();

      // psi @ Q^T (complex) via 4 real WMMA chains, double-buffered A frags
      const _Float16* qr = Qf + (size_t)(j * 2 + 0) * 16384;
      const _Float16* qi = Qf + (size_t)(j * 2 + 1) * 16384;
      v8f arr, aii, ari;
#pragma unroll
      for (int i = 0; i < 8; ++i) { arr[i] = 0.f; aii[i] = 0.f; ari[i] = 0.f; }
      {
        v16h arB = load_a_frag(&FR[0][0], 136, lane, 0);
        v16h aiB = load_a_frag(&FI[0][0], 136, lane, 0);
#pragma unroll
        for (int ks = 0; ks < 4; ++ks) {
          v16h ar = arB, ai = aiB;
          if (ks < 3) {
            arB = load_a_frag(&FR[0][0], 136, lane, (ks + 1) * 32);
            aiB = load_a_frag(&FI[0][0], 136, lane, (ks + 1) * 32);
          }
          v16h br = *(const v16h*)(qr + ((size_t)(wave * 4 + ks) * 512 + lane * 16));
          v16h bi = *(const v16h*)(qi + ((size_t)(wave * 4 + ks) * 512 + lane * 16));
          arr = wmma_f16(ar, br, arr);
          aii = wmma_f16(ai, bi, aii);
          ari = wmma_f16(ar, bi, ari);
          ari = wmma_f16(ai, br, ari);
        }
      }
#pragma unroll
      for (int r = 0; r < 8; ++r) {
        int m = r + 8 * hi;
        QR[m][nn] = arr[r] - aii[r];
        QI[m][nn] = ari[r];
      }
      __syncthreads();

      // single-qubit gate G = RY@RX on each of the 7 qubits
      float tx = coeffs[2 * j], ty = coeffs[2 * j + 1];
      float cx = cosf(tx * 0.5f), sx = sinf(tx * 0.5f);
      float cy = cosf(ty * 0.5f), sy = sinf(ty * 0.5f);
      float g00r = cy * cx, g00i = sy * sx;
      float g01r = -sy * cx, g01i = -cy * sx;
      float g10r = sy * cx, g10i = -cy * sx;
      float g11r = cy * cx, g11i = -sy * sx;
      for (int q = 0; q < 7; ++q) {
        int bpos = 6 - q, bit = 1 << bpos, mask = bit - 1;
        for (int i = tid; i < 16 * 64; i += 256) {
          int m = i >> 6, pr = i & 63;
          int s0 = ((pr & ~mask) << 1) | (pr & mask);
          int s1 = s0 | bit;
          float ar0 = QR[m][s0], ai0 = QI[m][s0];
          float ar1 = QR[m][s1], ai1 = QI[m][s1];
          float nar = g00r * ar0 - g00i * ai0 + g01r * ar1 - g01i * ai1;
          float nai = g00r * ai0 + g00i * ar0 + g01r * ai1 + g01i * ar1;
          float nbr = g10r * ar0 - g10i * ai0 + g11r * ar1 - g11i * ai1;
          float nbi = g10r * ai0 + g10i * ar0 + g11r * ai1 + g11i * ar1;
          QR[m][s0] = nar; QI[m][s0] = nai;
          QR[m][s1] = nbr; QI[m][s1] = nbi;
        }
        __syncthreads();
      }

      // CNOT-ring permutation: new[:, d] = old[:, idx(d)]
      for (int i = tid; i < 2048; i += 256) {
        int m = i >> 7, d = i & 127;
        int idx = d;
        if (idx & 1) idx ^= 64;  // reversed gate (6,0)
        for (int c = 5; c >= 0; --c) {
          int cbit = 1 << (6 - c), tbit = 1 << (5 - c);
          if (idx & cbit) idx ^= tbit;
        }
        PR[m][d] = QR[m][idx];
        PI[m][d] = QI[m][idx];
      }
      __syncthreads();
    }

    // decoder: out = h + real(psi) @ W_dec + b_dec
    for (int i = tid; i < 2048; i += 256) {
      int m = i >> 7, k = i & 127;
      FR[m][k] = (_Float16)PR[m][k];
    }
    __syncthreads();
    v8f accd;
#pragma unroll
    for (int i = 0; i < 8; ++i) accd[i] = 0.f;
    {
      v16h a = load_a_frag(&FR[0][0], 136, lane, 0);
#pragma unroll
      for (int ks = 0; ks < 4; ++ks) {
        v16h cur = a;
        if (ks < 3) a = load_a_frag(&FR[0][0], 136, lane, (ks + 1) * 32);
        v16h b = *(const v16h*)(Wdecf + ((size_t)(wave * 4 + ks) * 512 + lane * 16));
        accd = wmma_f16(cur, b, accd);
      }
    }
#pragma unroll
    for (int r = 0; r < 8; ++r) {
      int m = r + 8 * hi;
      int node = t * 16 + m;
      if (node < N)
        out[(size_t)node * 128 + nn] = h[(size_t)node * 128 + nn] + accd[r] + bdec;
    }
    __syncthreads();
  }
}

// ---------------------------------------------------------------------------
// Launch
// ---------------------------------------------------------------------------
extern "C" void kernel_launch(void* const* d_in, const int* in_sizes, int n_in,
                              void* d_out, int out_size, void* d_ws, size_t ws_size,
                              hipStream_t stream) {
  const float* h      = (const float*)d_in[0];
  const int*   ei     = (const int*)d_in[1];
  const float* W_e1   = (const float*)d_in[2];
  const float* b_e1   = (const float*)d_in[3];
  const float* W_e2   = (const float*)d_in[4];
  const float* b_e2   = (const float*)d_in[5];
  const float* W_enc  = (const float*)d_in[6];
  const float* b_enc  = (const float*)d_in[7];
  const float* coeffs = (const float*)d_in[8];
  const float* A_real = (const float*)d_in[9];
  const float* A_imag = (const float*)d_in[10];
  const float* W_dec  = (const float*)d_in[11];
  const float* b_dec  = (const float*)d_in[12];

  const int N = in_sizes[0] / 128;
  const int E = in_sizes[1] / 2;

  char* ws = (char*)d_ws;
  _Float16* We1f  = (_Float16*)(ws + 0);        //  64 KB
  _Float16* We2f  = (_Float16*)(ws + 65536);    //  32 KB
  _Float16* Wencf = (_Float16*)(ws + 98304);    //  64 KB
  _Float16* Wdecf = (_Float16*)(ws + 163840);   //  32 KB
  _Float16* Qf    = (_Float16*)(ws + 196608);   // 128 KB (2 layers x r/i frags)
  float*    aug   = (float*)(ws + 327680);      // 512 KB (GJ augmented, 2 layers)
  float*    Qt    = (float*)(ws + 851968);      // 512 KB (Q^T r/i planes)
  float*    agg   = (float*)(ws + 1376256);     // N*128*4 bytes

  float* out = (float*)d_out;
  float* edge_feat = out + (size_t)N * 128;

  pack_b_frags<<<(256 * 128 + 255) / 256, 256, 0, stream>>>(W_e1, We1f, 256);
  pack_b_frags<<<(128 * 128 + 255) / 256, 256, 0, stream>>>(W_e2, We2f, 128);
  pack_b_frags<<<(256 * 128 + 255) / 256, 256, 0, stream>>>(W_enc, Wencf, 256);
  pack_b_frags<<<(128 * 128 + 255) / 256, 256, 0, stream>>>(W_dec, Wdecf, 128);

  cayley_kernel<<<2, 256, 0, stream>>>(A_real, A_imag, aug, Qt);
  for (int j = 0; j < 2; ++j)
    for (int part = 0; part < 2; ++part)
      pack_b_frags<<<(128 * 128 + 255) / 256, 256, 0, stream>>>(
          Qt + (size_t)(j * 2 + part) * 16384, Qf + (size_t)(j * 2 + part) * 16384, 128);

  long aggN = (long)N * 128;
  zero_f32<<<(int)((aggN + 255) / 256), 256, 0, stream>>>(agg, aggN);

  edge_kernel<<<2048, 256, 0, stream>>>(h, ei, We1f, We2f, b_e1, b_e2, agg,
                                        edge_feat, N, E);
  node_kernel<<<1024, 256, 0, stream>>>(h, agg, Wencf, Wdecf, Qf, b_enc, b_dec,
                                        coeffs, out, N);
}
